// Seq2Seq_51470888075756
// MI455X (gfx1250) — matrix operand order repaired and checked by reading, not compiled, not run on hardware
//
#include <hip/hip_runtime.h>

// ---------------------------------------------------------------------------
// Seq2Seq LSTM (encoder 256 steps, decoder 128 steps) for MI455X / gfx1250.
//
// Parallel decomposition:
//   - batch rows are independent through the recurrence -> 128 blocks, each
//     owning 16 batch rows (one WMMA M-tile) for the whole sequence.
//   - within a block, 4 waves split the 512 gate outputs: wave ws computes
//     hidden-column tiles j = {2ws, 2ws+1} for all four gates (i,f,g,o live
//     at N-tiles j, j+8, j+16, j+24), so the pointwise LSTM update is
//     wave-local. h is exchanged via a ping-pong LDS buffer + one barrier
//     per time step.
//   - weights are pre-packed into exact WMMA B-fragment layout (bf16) and
//     staged in LDS (~220 KB of the 320 KB/WGP), via async global->LDS
//     loads (GLOBAL_LOAD_ASYNC_TO_LDS_B128, ASYNCcnt-tracked).
// Math: v_wmma_f32_16x16x32_bf16 (bf16 in, f32 accumulate), f32 pointwise
// with exp-based sigmoid/tanh, f32 cell state in registers.
// ---------------------------------------------------------------------------

typedef __attribute__((ext_vector_type(16))) __bf16 v16bf;
typedef __attribute__((ext_vector_type(8)))  float  v8f;
typedef __attribute__((ext_vector_type(4)))  int    v4i;

#define SEQ_LEN 256
#define BATCH   2048
#define N_IN    64
#define N_OUT   64
#define EMB     128
#define OUT_LEN 128

// packed-weight dword offsets inside workspace
#define OFF_ENC_WIH 0
#define OFF_ENC_WHH 16384
#define OFF_DEC_WIH 49152
#define OFF_DEC_WHH 65536
#define OFF_OUT_W   98304
// total packed dwords = 102400 (400 KB of d_ws)

#if defined(__HIP_DEVICE_COMPILE__) &&                                        \
    __has_builtin(__builtin_amdgcn_global_load_async_to_lds_b128) &&          \
    __has_builtin(__builtin_amdgcn_s_wait_asynccnt)
#define USE_ASYNC_LDS 1
// pointee types for the async builtin: v4i in global (AS1) / LDS (AS3)
typedef __attribute__((address_space(1))) v4i g_v4i;
typedef __attribute__((address_space(3))) v4i l_v4i;
#else
#define USE_ASYNC_LDS 0
#endif

__device__ __forceinline__ unsigned short f2bf(float f) {
  unsigned int u = __float_as_uint(f);
  u += 0x7FFFu + ((u >> 16) & 1u);          // round-to-nearest-even
  return (unsigned short)(u >> 16);
}

union BF16x16 { unsigned int u[8]; uint4 q[2]; v16bf v; };
union F32x8   { float f[8]; v8f v; };

__device__ __forceinline__ v8f splat8(float x) {
  F32x8 t;
#pragma unroll
  for (int i = 0; i < 8; ++i) t.f[i] = x;
  return t.v;
}

__device__ __forceinline__ float sigmoidf_(float x) {
  return 1.0f / (1.0f + __expf(-x));        // v_exp_f32 + v_rcp_f32 path
}
__device__ __forceinline__ float tanh_fast(float x) {
  return 2.0f * sigmoidf_(2.0f * x) - 1.0f; // avoids libm tanhf expansion
}

__device__ __forceinline__ v8f wmma_bf16(v16bf a, v16bf b, v8f c) {
  // D = A(16x32 bf16) * B(32x16 bf16) + C(16x16 f32)
  return __builtin_amdgcn_wmma_f32_16x16x32_bf16(false, a, false, b,
                                                 (short)0, c, false, false);
}

// B fragment, pre-packed lane-major: lane reads 8 contiguous dwords.
__device__ __forceinline__ v16bf load_bfrag(const unsigned int* fragbase, int lane) {
  BF16x16 r;
  const uint4* p = (const uint4*)(fragbase + lane * 8);
  r.q[0] = p[0];
  r.q[1] = p[1];
  return r.v;
}

// A fragment from global f32 row (this lane's batch row).
// Lane layout (16-bit A 16x32): row = lane&15, K runs {k1..k1+7, k1+16..k1+23}
// with k1 = 32*kt + 8*(lane>=16). VGPR0-3 = first run, VGPR4-7 = second run.
__device__ __forceinline__ v16bf load_afrag_global(const float* rowbase, int k1) {
  BF16x16 r;
#pragma unroll
  for (int h = 0; h < 2; ++h) {
    const float4* p = (const float4*)(rowbase + k1 + 16 * h);
    float4 a = p[0], b = p[1];
    r.u[4 * h + 0] = (unsigned)f2bf(a.x) | ((unsigned)f2bf(a.y) << 16);
    r.u[4 * h + 1] = (unsigned)f2bf(a.z) | ((unsigned)f2bf(a.w) << 16);
    r.u[4 * h + 2] = (unsigned)f2bf(b.x) | ((unsigned)f2bf(b.y) << 16);
    r.u[4 * h + 3] = (unsigned)f2bf(b.z) | ((unsigned)f2bf(b.w) << 16);
  }
  return r.v;
}

// A fragment from bf16 LDS staging row (row-major, row = lane&15).
__device__ __forceinline__ v16bf load_afrag_lds(const unsigned short* rowbase, int k1) {
  BF16x16 r;
  r.q[0] = *(const uint4*)(rowbase + k1);
  r.q[1] = *(const uint4*)(rowbase + k1 + 16);
  return r.v;
}

// ---------------------------------------------------------------------------
// Prep kernel: pack W ([Nout][K] row-major f32, gates = x @ W^T) into the
// per-lane WMMA B-fragment layout: frag(kt,j), dword(lane,v) =
//   bf16(W[16j + lane%16][32kt + 16*(v/4) + 2*(v%4) + 8*(lane/16)]) | bf16(...k+1) << 16
// ---------------------------------------------------------------------------
__global__ void pack_bmat(const float* __restrict__ W, int K, int jtiles,
                          unsigned int* __restrict__ dst) {
  int idx = blockIdx.x * blockDim.x + threadIdx.x;
  int total = (K / 32) * jtiles * 256;
  if (idx >= total) return;
  int frag = idx >> 8;
  int d    = idx & 255;
  int l    = d >> 3;          // lane 0..31
  int v    = d & 7;           // vgpr 0..7
  int kt   = frag / jtiles;
  int j    = frag % jtiles;
  int n    = 16 * j + (l & 15);
  int k    = 32 * kt + 16 * (v >> 2) + 2 * (v & 3) + 8 * (l >> 4);
  unsigned int lo = f2bf(W[(size_t)n * K + k]);
  unsigned int hi = f2bf(W[(size_t)n * K + k + 1]);
  dst[(size_t)frag * 256 + l * 8 + v] = lo | (hi << 16);
}

// ---------------------------------------------------------------------------
// Main persistent kernel
// ---------------------------------------------------------------------------
struct __align__(16) SMem {
  unsigned int wih[16384];        // 64 KB  : packed Wih (enc, then dec)
  unsigned int whh[32768];        // 128 KB : packed Whh (enc, then dec)
  unsigned int wout[4096];        // 16 KB  : packed out_W (decoder phase)
  float bias[512];                // gate bias
  float outb[64];                 // output bias
  unsigned short hstage[2][16 * EMB];  // ping-pong h exchange (bf16)
  unsigned short ystage[16 * N_OUT];   // decoder y feedback (bf16)
};

// Stage a packed-weight region into LDS. Uses CDNA5 async global->LDS
// (GLOBAL_LOAD_ASYNC_TO_LDS_B128, ASYNCcnt) when available; caller issues
// stage_weights_wait() + __syncthreads() before consuming.
__device__ __forceinline__ void stage_weights(unsigned int* dst,
                                              const unsigned int* src,
                                              int ndw, int tid) {
#if USE_ASYNC_LDS
  for (int i = tid; i < (ndw >> 2); i += 128)
    __builtin_amdgcn_global_load_async_to_lds_b128(
        (g_v4i*)(src + 4 * (size_t)i),
        (l_v4i*)(dst + 4 * (size_t)i),
        0, 0);
#else
  for (int i = tid; i < (ndw >> 2); i += 128)
    ((uint4*)dst)[i] = ((const uint4*)src)[i];
#endif
}

__device__ __forceinline__ void stage_weights_wait() {
#if USE_ASYNC_LDS
  __builtin_amdgcn_s_wait_asynccnt(0);
#endif
}

// Gate GEMM + pointwise update for this wave's 2 hidden-column tiles
// (j = jbase, jbase+1). Writes h_new columns into hbuf (shared, disjoint
// per wave). Caller must barrier before re-reading hbuf.
__device__ __forceinline__ void lstm_gates_update(
    const v16bf xfrag[2], const v16bf hfrag[4], v8f cfrag[2],
    const unsigned int* wih, const unsigned int* whh, const float* bias,
    unsigned short* hbuf, int jbase, int lane, int lmod, int lhalf) {
#pragma unroll 1
  for (int jj = 0; jj < 2; ++jj) {
    const int j = jbase + jj;
    v8f acc[4];                       // i, f, g, o tiles (j, j+8, j+16, j+24)
#pragma unroll
    for (int g = 0; g < 4; ++g) acc[g] = splat8(bias[16 * (j + 8 * g) + lmod]);
#pragma unroll
    for (int g = 0; g < 4; ++g) {
      const int nt = j + 8 * g;
#pragma unroll
      for (int kt = 0; kt < 2; ++kt)  // x @ Wih^T  (K = 64)
        acc[g] = wmma_bf16(xfrag[kt], load_bfrag(wih + (kt * 32 + nt) * 256, lane), acc[g]);
#pragma unroll
      for (int kt = 0; kt < 4; ++kt)  // h @ Whh^T  (K = 128)
        acc[g] = wmma_bf16(hfrag[kt], load_bfrag(whh + (kt * 32 + nt) * 256, lane), acc[g]);
    }
    // pointwise; D-layout: vgpr r -> row r+8*lhalf, lane -> col lmod
    F32x8 gi, gf, gg, go, cc;
    gi.v = acc[0]; gf.v = acc[1]; gg.v = acc[2]; go.v = acc[3];
    cc.v = cfrag[jj];
#pragma unroll
    for (int r = 0; r < 8; ++r) {
      float iv = sigmoidf_(gi.f[r]);
      float fv = sigmoidf_(gf.f[r]);
      float gv = tanh_fast(gg.f[r]);
      float ov = sigmoidf_(go.f[r]);
      float cn = fv * cc.f[r] + iv * gv;
      cc.f[r] = cn;
      float hn = ov * tanh_fast(cn);
      hbuf[(r + 8 * lhalf) * EMB + 16 * j + lmod] = f2bf(hn);
    }
    cfrag[jj] = cc.v;
  }
}

__global__ void __launch_bounds__(128, 1)
seq2seq_kernel(const float* __restrict__ x,
               const float* __restrict__ enc_b,
               const float* __restrict__ dec_b,
               const float* __restrict__ out_bias,
               const unsigned int* __restrict__ wp,
               float* __restrict__ out) {
  __shared__ SMem sm;
  const int tid   = threadIdx.x;
  const int ws    = tid >> 5;          // wave index = gate-column split
  const int lane  = tid & 31;
  const int lmod  = lane & 15;
  const int lhalf = lane >> 4;
  const int jbase = 2 * ws;
  const int brow  = blockIdx.x * 16;   // this block's 16 batch rows

  // ---- stage encoder weights ----
  stage_weights(sm.wih, wp + OFF_ENC_WIH, 16384, tid);
  stage_weights(sm.whh, wp + OFF_ENC_WHH, 32768, tid);
  for (int i = tid; i < 512; i += 128) sm.bias[i] = enc_b[i];
  stage_weights_wait();
  __syncthreads();

  v16bf hfrag[4];
  v8f   cfrag[2];
  {
    BF16x16 z;
#pragma unroll
    for (int i = 0; i < 8; ++i) z.u[i] = 0;
#pragma unroll
    for (int i = 0; i < 4; ++i) hfrag[i] = z.v;
    cfrag[0] = splat8(0.0f);
    cfrag[1] = splat8(0.0f);
  }

  // ================= encoder =================
#pragma unroll 1
  for (int t = 0; t < SEQ_LEN; ++t) {
    const float* xrow = x + ((size_t)t * BATCH + brow + lmod) * N_IN;
    if (t + 1 < SEQ_LEN)               // stream next step's tile into caches
      __builtin_prefetch(xrow + (size_t)BATCH * N_IN, 0, 0);
    v16bf xfrag[2];
#pragma unroll
    for (int kt = 0; kt < 2; ++kt)
      xfrag[kt] = load_afrag_global(xrow, 32 * kt + 8 * lhalf);

    unsigned short* hbuf = &sm.hstage[t & 1][0];
    lstm_gates_update(xfrag, hfrag, cfrag, sm.wih, sm.whh, sm.bias,
                      hbuf, jbase, lane, lmod, lhalf);
    __syncthreads();                   // h columns from all 4 waves visible
#pragma unroll
    for (int kt = 0; kt < 4; ++kt)
      hfrag[kt] = load_afrag_lds(hbuf + lmod * EMB, 32 * kt + 8 * lhalf);
  }

  // ---- swap in decoder weights (h,c carry over from encoder) ----
  __syncthreads();
  stage_weights(sm.wih,  wp + OFF_DEC_WIH, 16384, tid);
  stage_weights(sm.whh,  wp + OFF_DEC_WHH, 32768, tid);
  stage_weights(sm.wout, wp + OFF_OUT_W,    4096, tid);
  for (int i = tid; i < 512; i += 128) sm.bias[i] = dec_b[i];
  if (tid < 64) sm.outb[tid] = out_bias[tid];
  for (int i = tid; i < 16 * N_OUT; i += 128) sm.ystage[i] = 0;  // x0 = 0
  stage_weights_wait();
  __syncthreads();

  // ================= decoder (autoregressive) =================
#pragma unroll 1
  for (int t = 0; t < OUT_LEN; ++t) {
    v16bf xfrag[2];
#pragma unroll
    for (int kt = 0; kt < 2; ++kt)
      xfrag[kt] = load_afrag_lds(&sm.ystage[lmod * N_OUT], 32 * kt + 8 * lhalf);

    unsigned short* hbuf = &sm.hstage[t & 1][0];
    lstm_gates_update(xfrag, hfrag, cfrag, sm.wih, sm.whh, sm.bias,
                      hbuf, jbase, lane, lmod, lhalf);
    __syncthreads();                   // h complete; also: all y reads done
#pragma unroll
    for (int kt = 0; kt < 4; ++kt)
      hfrag[kt] = load_afrag_lds(hbuf + lmod * EMB, 32 * kt + 8 * lhalf);

    // y = h_new @ out_W^T + out_b ; wave ws computes y-tile j = ws
    {
      v8f acc = splat8(sm.outb[16 * ws + lmod]);
#pragma unroll
      for (int kt = 0; kt < 4; ++kt)
        acc = wmma_bf16(hfrag[kt], load_bfrag(sm.wout + (kt * 4 + ws) * 256, lane), acc);
      F32x8 yy; yy.v = acc;
#pragma unroll
      for (int r = 0; r < 8; ++r) {
        int row = r + 8 * lhalf;
        int m   = 16 * ws + lmod;
        sm.ystage[row * N_OUT + m] = f2bf(yy.f[r]);
        out[(size_t)(brow + row) * (N_OUT * OUT_LEN) + (size_t)m * OUT_LEN + t] = yy.f[r];
      }
    }
    __syncthreads();                   // y feedback visible before next read
  }
}

// ---------------------------------------------------------------------------
extern "C" void kernel_launch(void* const* d_in, const int* in_sizes, int n_in,
                              void* d_out, int out_size, void* d_ws, size_t ws_size,
                              hipStream_t stream) {
  const float* x       = (const float*)d_in[0];
  const float* enc_Wih = (const float*)d_in[1];
  const float* enc_Whh = (const float*)d_in[2];
  const float* enc_b   = (const float*)d_in[3];
  const float* dec_Wih = (const float*)d_in[4];
  const float* dec_Whh = (const float*)d_in[5];
  const float* dec_b   = (const float*)d_in[6];
  const float* out_W   = (const float*)d_in[7];
  const float* out_b   = (const float*)d_in[8];
  float* out           = (float*)d_out;
  unsigned int* wp     = (unsigned int*)d_ws;

  // pack all weight matrices into WMMA B-fragment layout (bf16)
  pack_bmat<<<64,  256, 0, stream>>>(enc_Wih,  64, 32, wp + OFF_ENC_WIH);
  pack_bmat<<<128, 256, 0, stream>>>(enc_Whh, 128, 32, wp + OFF_ENC_WHH);
  pack_bmat<<<64,  256, 0, stream>>>(dec_Wih,  64, 32, wp + OFF_DEC_WIH);
  pack_bmat<<<128, 256, 0, stream>>>(dec_Whh, 128, 32, wp + OFF_DEC_WHH);
  pack_bmat<<<16,  256, 0, stream>>>(out_W,   128,  4, wp + OFF_OUT_W);

  // 128 blocks x 4 waves; block owns 16 batch rows, waves split gate columns
  seq2seq_kernel<<<128, 128, 0, stream>>>(x, enc_b, dec_b, out_b, wp, out);
}
